// decompress_jpeg_24661702214017
// MI455X (gfx1250) — compile-verified, matrix operand order checked by reading, not run on hardware
//
#include <hip/hip_runtime.h>

typedef __attribute__((ext_vector_type(16))) _Float16 v16h;
typedef __attribute__((ext_vector_type(8)))  _Float16 v8h;
typedef __attribute__((ext_vector_type(8)))  float    v8f;
typedef __attribute__((ext_vector_type(4)))  float    f4;

union V16 { v16h v; v8h h[2]; };

// ---------------------------------------------------------------------------
// Kernel 0: build the folded 64x64 IDCT transforms (f16).
// T[m][k] = 0.25 * idct[x][y][u][v] * alpha[u][v] * table[u][v] * factor
// with m = x*8+y (output pixel), k = u*8+v (coefficient).
// idct_tensor flat index [x][y][u][v] == m*64 + k.
// ---------------------------------------------------------------------------
__global__ void prep_tables(const float* __restrict__ yt, const float* __restrict__ ct,
                            const float* __restrict__ alpha, const float* __restrict__ idct,
                            const int* __restrict__ factor,
                            _Float16* __restrict__ Ty, _Float16* __restrict__ Tc) {
    float f = (float)(*factor);
    for (int idx = threadIdx.x; idx < 4096; idx += blockDim.x) {
        int m = idx >> 6, k = idx & 63;
        float base = 0.25f * idct[m * 64 + k] * alpha[k] * f;
        Ty[idx] = (_Float16)(base * yt[k]);
        Tc[idx] = (_Float16)(base * ct[k]);
    }
}

// Load B-matrix fragment (K=32 x N=16, f16) for one k-tile of a 16-block batch.
// Lane n = lane&15 owns block column n; lanes<16 take k = c0..c0+7, c0+16..c0+23,
// lanes>=16 take the +8 variants (folded into c0).
static __device__ __forceinline__ v16h load_b(const float* __restrict__ bp, int c0) {
    const f4* p0 = (const f4*)(bp + c0);
    f4 f0 = p0[0], f1 = p0[1];
    const f4* p1 = (const f4*)(bp + c0 + 16);
    f4 f2 = p1[0], f3 = p1[1];
    V16 r;
#pragma unroll
    for (int i = 0; i < 4; ++i) {
        r.h[0][i]     = (_Float16)f0[i];
        r.h[0][i + 4] = (_Float16)f1[i];
        r.h[1][i]     = (_Float16)f2[i];
        r.h[1][i + 4] = (_Float16)f3[i];
    }
    return r.v;
}

// Load A-matrix fragment (M=16 x K=32, f16) for (Mtile, ktile) from LDS copy of T.
static __device__ __forceinline__ v16h load_a(const _Float16* sT, int Mt, int kt, int lane) {
    int row   = Mt * 16 + (lane & 15);
    int cbase = kt * 32 + ((lane >> 4) << 3);
    V16 r;
    r.h[0] = *(const v8h*)(sT + row * 64 + cbase);
    r.h[1] = *(const v8h*)(sT + row * 64 + cbase + 16);
    return r.v;
}

// ---------------------------------------------------------------------------
// Fused kernel: one workgroup (6 waves) owns a 16x256-pixel region:
//   2 Y-block rows x 32 Y-block cols  (waves 0-3, 16 blocks each)
//   1 chroma-block row x 16 chroma-block cols for Cb (wave 4) and Cr (wave 5)
// Chroma IDCT results stay in LDS; the epilogue fuses 2x upsample +
// color matrix + clip, so no intermediate planes ever hit HBM.
// grid = 16 images * 64 regRows * 4 regCols = 4096 workgroups.
// ---------------------------------------------------------------------------
__global__ void __launch_bounds__(192) jpeg_fused(const float* __restrict__ yIn,
                                                  const float* __restrict__ cbIn,
                                                  const float* __restrict__ crIn,
                                                  const _Float16* __restrict__ Ty,
                                                  const _Float16* __restrict__ Tc,
                                                  const float* __restrict__ matrix,
                                                  const float* __restrict__ shift,
                                                  float* __restrict__ out) {
    __shared__ __align__(16) _Float16 sTy[4096];
    __shared__ __align__(16) _Float16 sTc[4096];
    __shared__ __align__(16) float    sCb[8 * 128];
    __shared__ __align__(16) float    sCr[8 * 128];

    {
        const f4* sy = (const f4*)Ty;
        const f4* sc = (const f4*)Tc;
        f4* dy = (f4*)sTy;
        f4* dc = (f4*)sTc;
        for (int k = threadIdx.x; k < 512; k += 192) { dy[k] = sy[k]; dc[k] = sc[k]; }
    }

    const int r      = blockIdx.x;
    const int b      = r >> 8;        // image
    const int rr     = r & 255;
    const int regRow = rr >> 2;       // 0..63  (pair of Y block-rows)
    const int regCol = rr & 3;        // 0..3   (group of 32 Y block-cols)

    const int wave = __builtin_amdgcn_readfirstlane(threadIdx.x >> 5); // 0..5, wave-uniform
    const int lane = threadIdx.x & 31;
    const int n    = lane & 15;       // block column within the 16-block batch
    const int xoff = lane >> 4;       // 0/1: which row of the M pair this lane holds
    const int c0   = xoff << 3;

    // matrix[c*3+d] (already transposed by setup): rgb[d] = sum_c t[c]*matrix[c*3+d]
    const float m00 = matrix[0], m01 = matrix[1], m02 = matrix[2];
    const float m10 = matrix[3], m11 = matrix[4], m12 = matrix[5];
    const float m20 = matrix[6], m21 = matrix[7], m22 = matrix[8];
    // WMMA outputs exclude the reference's +128 bias; re-add here (chroma's cancels shift).
    const float s0 = shift[0] + 128.0f;
    const float s1 = shift[1] + 128.0f;
    const float s2 = shift[2] + 128.0f;

    __syncthreads();   // tables ready

    v8f acc[4] = {};
    if (wave < 4) {
        // ---- Y IDCT: 16 blocks, accumulators kept in registers ----
        const int i  = regRow * 2 + (wave >> 1);           // Y block-row 0..127
        const int j  = regCol * 32 + (wave & 1) * 16 + n;  // Y block-col 0..127
        const int nb = i * 128 + j;
        const float* bp = yIn + ((size_t)b * 16384 + nb) * 64;
        v16h b0 = load_b(bp, c0);
        v16h b1 = load_b(bp, 32 + c0);
#pragma unroll
        for (int Mt = 0; Mt < 4; ++Mt) {
            v16h a0 = load_a(sTy, Mt, 0, lane);
            v16h a1 = load_a(sTy, Mt, 1, lane);
            v8f c = {};
            c = __builtin_amdgcn_wmma_f32_16x16x32_f16(false, a0, false, b0, (short)0, c, false, false);
            c = __builtin_amdgcn_wmma_f32_16x16x32_f16(false, a1, false, b1, (short)0, c, false, false);
            acc[Mt] = c;
        }
    } else {
        // ---- chroma IDCT: wave 4 = Cb, wave 5 = Cr; results -> LDS tile [8][128] ----
        const float* src = (wave == 4) ? cbIn : crIn;
        float* sDst      = (wave == 4) ? sCb : sCr;
        const int ci = regRow;                 // chroma block-row 0..63
        const int cj = regCol * 16 + n;        // chroma block-col 0..63
        const int nb = ci * 64 + cj;
        const float* bp = src + ((size_t)b * 4096 + nb) * 64;
        v16h b0 = load_b(bp, c0);
        v16h b1 = load_b(bp, 32 + c0);
#pragma unroll
        for (int Mt = 0; Mt < 4; ++Mt) {
            v16h a0 = load_a(sTc, Mt, 0, lane);
            v16h a1 = load_a(sTc, Mt, 1, lane);
            v8f c = {};
            c = __builtin_amdgcn_wmma_f32_16x16x32_f16(false, a0, false, b0, (short)0, c, false, false);
            c = __builtin_amdgcn_wmma_f32_16x16x32_f16(false, a1, false, b1, (short)0, c, false, false);
            const int x = Mt * 2 + xoff;       // chroma row within block, 0..7
            float* rp = sDst + x * 128 + n * 8;
            f4 lo = { c[0], c[1], c[2], c[3] };
            f4 hi = { c[4], c[5], c[6], c[7] };
            *(f4*)rp       = lo;
            *(f4*)(rp + 4) = hi;
        }
    }

    __syncthreads();   // chroma tiles ready

    if (wave < 4) {
        // ---- fused upsample + color convert + clip epilogue ----
        const int i  = regRow * 2 + (wave >> 1);
        const int j  = regCol * 32 + (wave & 1) * 16 + n;
        const int py = j * 8;
        const int ccol = ((wave & 1) * 16 + n) * 4;   // region-local chroma col base
        float* outR = out + ((size_t)(b * 3 + 0) << 20) + py;
        float* outG = out + ((size_t)(b * 3 + 1) << 20) + py;
        float* outB = out + ((size_t)(b * 3 + 2) << 20) + py;
        const float inv255 = 1.0f / 255.0f;

#pragma unroll
        for (int Mt = 0; Mt < 4; ++Mt) {
            const v8f c  = acc[Mt];
            const int x  = Mt * 2 + xoff;             // row within Y block
            const int px = i * 8 + x;                 // global Y row
            const int crow = ((wave >> 1) * 8 + x) >> 1;  // region-local chroma row 0..7
            f4 cb4 = *(const f4*)(sCb + crow * 128 + ccol);
            f4 cr4 = *(const f4*)(sCr + crow * 128 + ccol);

            f4 rlo, rhi, glo, ghi, blo, bhi;
#pragma unroll
            for (int t = 0; t < 8; ++t) {
                float yv  = c[t]        + s0;
                float cbv = cb4[t >> 1] + s1;
                float crv = cr4[t >> 1] + s2;
                float rv = yv * m00 + cbv * m10 + crv * m20;
                float gv = yv * m01 + cbv * m11 + crv * m21;
                float bv = yv * m02 + cbv * m12 + crv * m22;
                rv = fminf(fmaxf(rv, 0.0f), 255.0f) * inv255;
                gv = fminf(fmaxf(gv, 0.0f), 255.0f) * inv255;
                bv = fminf(fmaxf(bv, 0.0f), 255.0f) * inv255;
                if (t < 4) { rlo[t] = rv; glo[t] = gv; blo[t] = bv; }
                else       { rhi[t - 4] = rv; ghi[t - 4] = gv; bhi[t - 4] = bv; }
            }
            f4* rrow = (f4*)(outR + (size_t)px * 1024);
            f4* grow = (f4*)(outG + (size_t)px * 1024);
            f4* brow = (f4*)(outB + (size_t)px * 1024);
            rrow[0] = rlo; rrow[1] = rhi;
            grow[0] = glo; grow[1] = ghi;
            brow[0] = blo; brow[1] = bhi;
        }
    }
}

// ---------------------------------------------------------------------------
extern "C" void kernel_launch(void* const* d_in, const int* in_sizes, int n_in,
                              void* d_out, int out_size, void* d_ws, size_t ws_size,
                              hipStream_t stream) {
    const float* y_in   = (const float*)d_in[0];
    const float* cb_in  = (const float*)d_in[1];
    const float* cr_in  = (const float*)d_in[2];
    const int*   factor = (const int*)d_in[5];
    const float* y_tab  = (const float*)d_in[6];
    const float* c_tab  = (const float*)d_in[7];
    const float* alpha  = (const float*)d_in[8];
    const float* idct   = (const float*)d_in[9];
    const float* matrix = (const float*)d_in[10];
    const float* shift  = (const float*)d_in[11];

    char* ws = (char*)d_ws;
    _Float16* Ty = (_Float16*)ws;           // 8 KiB
    _Float16* Tc = (_Float16*)(ws + 8192);  // 8 KiB

    float* out = (float*)d_out;

    prep_tables<<<1, 256, 0, stream>>>(y_tab, c_tab, alpha, idct, factor, Ty, Tc);
    jpeg_fused<<<4096, 192, 0, stream>>>(y_in, cb_in, cr_in, Ty, Tc, matrix, shift, out);
}
